// GQAAttention_62620623176121
// MI455X (gfx1250) — matrix operand order repaired and checked by reading, not compiled, not run on hardware
//
#include <hip/hip_runtime.h>
#include <hip/hip_bf16.h>

typedef __attribute__((ext_vector_type(16))) __bf16 v16bf;
typedef __attribute__((ext_vector_type(8)))  __bf16 v8bf;
typedef __attribute__((ext_vector_type(8)))  float  v8f;
typedef __attribute__((ext_vector_type(4)))  unsigned int u32x4;
typedef __attribute__((ext_vector_type(8)))  int i32x8;
typedef __attribute__((ext_vector_type(4)))  int i32x4;

#define NH     16
#define NKV    4
#define HD     64
#define HID    1024
#define BATCH  2
#define SEQ    2048
#define MROWS  (BATCH*SEQ)          // 4096
#define KVDIM  (NKV*HD)             // 256

union ABf { v16bf v; v8bf h[2]; };

// 16x32 bf16 A-fragment, contiguous k order (B fragments use the same):
// lane L: row = L%16; element e -> K = (L/16)*16 + e
__device__ __forceinline__ v16bf load_a_frag(const __bf16* rowPtr, int half) {
    ABf u;
    u.h[0] = *(const v8bf*)(rowPtr + half * 16);
    u.h[1] = *(const v8bf*)(rowPtr + half * 16 + 8);
    return u.v;
}

__device__ __forceinline__ v8f wmma_bf16(v16bf a, v16bf b, v8f c) {
    return __builtin_amdgcn_wmma_f32_16x16x32_bf16(false, a, false, b, (short)0, c, false, false);
}

// ---- Tensor Data Mover: 2D bf16 tile (rows x cols) from global -> LDS ----
// D# built per CDNA5 ISA (group0/group1 bitfields).
// 6-arg builtin form (clang-23 / therock headers): (g0, g1, g2, g3, g4, cpol)
__device__ __forceinline__ void tdm_load_2d(const __bf16* gtile, unsigned ldsOff,
                                            int tileRows, int tileCols,
                                            unsigned long long strideElems,
                                            unsigned tensorD0, unsigned tensorD1) {
    unsigned long long ga = (unsigned long long)(size_t)gtile;
    u32x4 g0;
    g0[0] = 1u;                                              // count=1, user mode
    g0[1] = ldsOff;                                          // lds_addr (bytes)
    g0[2] = (unsigned)(ga & 0xffffffffull);                  // global_addr[31:0]
    g0[3] = (unsigned)((ga >> 32) & 0x1ffffffull) | (2u << 30); // addr[56:32] | type=2
    i32x8 g1;
    g1[0] = (int)(1u << 16);                                 // data_size=1 (2 bytes)
    g1[1] = (int)((tensorD0 & 0xffffu) << 16);               // tensor_dim0[15:0]
    g1[2] = (int)(((tensorD0 >> 16) & 0xffffu) | ((tensorD1 & 0xffffu) << 16));
    g1[3] = (int)(((tensorD1 >> 16) & 0xffffu) | (((unsigned)tileCols & 0xffffu) << 16)); // tile_dim0
    g1[4] = (int)((unsigned)tileRows & 0xffffu);             // tile_dim1 (tile_dim2=0)
    g1[5] = (int)(unsigned)(strideElems & 0xffffffffull);    // tensor_dim0_stride[31:0]
    g1[6] = (int)(unsigned)((strideElems >> 32) & 0xffffull);// stride[47:32], dim1_stride=0
    g1[7] = 0;
    i32x4 z4 = {0, 0, 0, 0};
    i32x8 z8 = {0, 0, 0, 0, 0, 0, 0, 0};
    __builtin_amdgcn_tensor_load_to_lds(g0, g1, z4, z4, z8, 0);
}

// ---------------- conversion / transpose kernels ----------------

__global__ void cvt_f32_to_bf16(const float* __restrict__ in, __bf16* __restrict__ out, int n) {
    int i = blockIdx.x * blockDim.x + threadIdx.x;
    if (i < n) out[i] = (__bf16)in[i];
}

// in: [Kd, Nd] f32 row-major -> out: [Nd, Kd] bf16
__global__ void cvt_transpose_bf16(const float* __restrict__ in, __bf16* __restrict__ out,
                                   int Kd, int Nd) {
    int i = blockIdx.x * blockDim.x + threadIdx.x;
    if (i < Kd * Nd) {
        int n = i / Kd, k = i % Kd;
        out[i] = (__bf16)in[(size_t)k * Nd + n];
    }
}

// v: [B*S, 256] bf16 -> vt: [b, kv, d, s] bf16
__global__ void transpose_v_kernel(const __bf16* __restrict__ v, __bf16* __restrict__ vt) {
    int i = blockIdx.x * blockDim.x + threadIdx.x;
    const int total = BATCH * NKV * HD * SEQ;
    if (i < total) {
        int s = i % SEQ;
        int rest = i / SEQ;
        int d = rest % HD;
        int bk = rest / HD;
        int b = bk / NKV, kv = bk % NKV;
        vt[i] = v[(size_t)(b * SEQ + s) * KVDIM + kv * HD + d];
    }
}

// ---------------- TDM-fed, LDS-tiled WMMA GEMM ----------------
// C[M,N] = A[M,K](bf16) * Bt[N,K](bf16) + bias
// Block: 256 threads (8 waves), computes 128x128 C tile.
// Wave (wm,wn) in 2x4 grid computes 64x32 (4x2 subtiles of 16x16).
// K loop step 32; A/B 128x32 panels double-buffered in LDS via TDM.

__global__ void gemm_bf16_tdm(const __bf16* __restrict__ A, const __bf16* __restrict__ Bt,
                              const float* __restrict__ bias,
                              __bf16* __restrict__ outBf, float* __restrict__ outF,
                              int Mn, int Nn, int Kn) {
    __shared__ __bf16 lA[2][128][32];
    __shared__ __bf16 lB[2][128][32];

    const int lane = threadIdx.x & 31;
    const int wave = threadIdx.x >> 5;
    const int half = lane >> 4, l16 = lane & 15;
    const int wm = wave >> 2;          // 0..1  (64-row band)
    const int wn = wave & 3;           // 0..3  (32-col band)

    const int tilesNb = Nn >> 7;
    const int bm = blockIdx.x / tilesNb;
    const int bn = blockIdx.x % tilesNb;

    const __bf16* aTile = A + (size_t)(bm * 128) * Kn;   // +k0 per step
    const __bf16* bTile = Bt + (size_t)(bn * 128) * Kn;

    v8f acc[4][2];
#pragma unroll
    for (int mi = 0; mi < 4; ++mi)
#pragma unroll
        for (int ni = 0; ni < 2; ++ni) acc[mi][ni] = (v8f){};

    // prologue: fill buffer 0
    if (wave == 0) {
        tdm_load_2d(aTile, (unsigned)(size_t)&lA[0][0][0], 128, 32,
                    (unsigned long long)Kn, (unsigned)Kn, (unsigned)Mn);
        tdm_load_2d(bTile, (unsigned)(size_t)&lB[0][0][0], 128, 32,
                    (unsigned long long)Kn, (unsigned)Kn, (unsigned)Nn);
        __builtin_amdgcn_s_wait_tensorcnt(0);
    }
    __syncthreads();

    int buf = 0;
    for (int k0 = 0; k0 < Kn; k0 += 32) {
        const int nbuf = buf ^ 1;
        if (wave == 0 && (k0 + 32) < Kn) {
            tdm_load_2d(aTile + (k0 + 32), (unsigned)(size_t)&lA[nbuf][0][0], 128, 32,
                        (unsigned long long)Kn, (unsigned)Kn, (unsigned)Mn);
            tdm_load_2d(bTile + (k0 + 32), (unsigned)(size_t)&lB[nbuf][0][0], 128, 32,
                        (unsigned long long)Kn, (unsigned)Kn, (unsigned)Nn);
        }

        // fragments from LDS
        v16bf af[4], bf[2];
#pragma unroll
        for (int mi = 0; mi < 4; ++mi)
            af[mi] = load_a_frag(&lA[buf][wm * 64 + mi * 16 + l16][0], half);
#pragma unroll
        for (int ni = 0; ni < 2; ++ni)
            bf[ni] = *(const v16bf*)(&lB[buf][wn * 32 + ni * 16 + l16][half * 16]);

#pragma unroll
        for (int mi = 0; mi < 4; ++mi)
#pragma unroll
            for (int ni = 0; ni < 2; ++ni)
                acc[mi][ni] = wmma_bf16(af[mi], bf[ni], acc[mi][ni]);

        if (wave == 0) __builtin_amdgcn_s_wait_tensorcnt(0);
        __syncthreads();
        buf = nbuf;
    }

    // epilogue
#pragma unroll
    for (int ni = 0; ni < 2; ++ni) {
        const int col = bn * 128 + wn * 32 + ni * 16 + l16;
        const float bv = bias ? bias[col] : 0.0f;
#pragma unroll
        for (int mi = 0; mi < 4; ++mi) {
#pragma unroll
            for (int r = 0; r < 8; ++r) {
                int row = bm * 128 + wm * 64 + mi * 16 + half * 8 + r;
                float val = acc[mi][ni][r] + bv;
                if (outF) outF[(size_t)row * Nn + col] = val;
                else      outBf[(size_t)row * Nn + col] = (__bf16)val;
            }
        }
    }
}

// ---------------- fused GQA flash attention (one wave per 16-query tile) ----------------

__global__ void gqa_attn_wmma(const __bf16* __restrict__ Q,   // [B*S, 1024]
                              const __bf16* __restrict__ Km,  // [B*S, 256]
                              const __bf16* __restrict__ Vt,  // [b, kv, d, s]
                              __bf16* __restrict__ O) {       // [B*S, 1024]
    __shared__ float pbuf[8][16][32];
    const int lane = threadIdx.x & 31;
    const int wave = threadIdx.x >> 5;
    const int g = blockIdx.x * (blockDim.x >> 5) + wave;
    const int qtPerHead = SEQ / 16;
    const int b  = g / (NH * qtPerHead);
    const int rem = g % (NH * qtPerHead);
    const int h  = rem / qtPerHead;
    const int qt = rem % qtPerHead;
    const int kv = h / (NH / NKV);
    const int half = lane >> 4, l16 = lane & 15;

    const __bf16* qRow = Q + (size_t)(b * SEQ + qt * 16 + l16) * HID + h * HD;
    const v16bf qa0 = load_a_frag(qRow,      half);
    const v16bf qa1 = load_a_frag(qRow + 32, half);

    float m[8], l[8];
#pragma unroll
    for (int r = 0; r < 8; ++r) { m[r] = -3.0e38f; l[r] = 0.0f; }
    v8f o0 = {}, o1 = {}, o2 = {}, o3 = {};

    const __bf16* kBase = Km + (size_t)b * SEQ * KVDIM + kv * HD;
    const __bf16* vBase = Vt + (size_t)(b * NKV + kv) * HD * SEQ;

    for (int kb = 0; kb < SEQ; kb += 32) {
        const __bf16* k0r = kBase + (size_t)(kb + l16) * KVDIM + half * 16;
        const __bf16* k1r = kBase + (size_t)(kb + 16 + l16) * KVDIM + half * 16;
        v8f s0 = {}, s1 = {};
        s0 = wmma_bf16(qa0, *(const v16bf*)(k0r),      s0);
        s0 = wmma_bf16(qa1, *(const v16bf*)(k0r + 32), s0);
        s1 = wmma_bf16(qa0, *(const v16bf*)(k1r),      s1);
        s1 = wmma_bf16(qa1, *(const v16bf*)(k1r + 32), s1);

#pragma unroll
        for (int r = 0; r < 8; ++r) {
            float mx = fmaxf(s0[r], s1[r]);
            mx = fmaxf(mx, __shfl_xor(mx, 1, 32));
            mx = fmaxf(mx, __shfl_xor(mx, 2, 32));
            mx = fmaxf(mx, __shfl_xor(mx, 4, 32));
            mx = fmaxf(mx, __shfl_xor(mx, 8, 32));
            float mn = fmaxf(m[r], mx);
            float alpha = __expf(m[r] - mn);
            m[r] = mn;
            float p0 = __expf(s0[r] - mn);
            float p1 = __expf(s1[r] - mn);
            float sum = p0 + p1;
            sum += __shfl_xor(sum, 1, 32);
            sum += __shfl_xor(sum, 2, 32);
            sum += __shfl_xor(sum, 4, 32);
            sum += __shfl_xor(sum, 8, 32);
            l[r] = l[r] * alpha + sum;
            o0[r] *= alpha; o1[r] *= alpha; o2[r] *= alpha; o3[r] *= alpha;
            pbuf[wave][half * 8 + r][l16]      = p0;
            pbuf[wave][half * 8 + r][16 + l16] = p1;
        }
        __asm__ volatile("s_wait_dscnt 0" ::: "memory");

        v16bf pa;
#pragma unroll
        for (int e = 0; e < 16; ++e) {
            int key = half * 16 + e;
            pa[e] = (__bf16)pbuf[wave][l16][key];
        }

        const __bf16* vcol = vBase + (size_t)l16 * SEQ + kb + half * 16;
        o0 = wmma_bf16(pa, *(const v16bf*)(vcol + (size_t)(0 * 16) * SEQ), o0);
        o1 = wmma_bf16(pa, *(const v16bf*)(vcol + (size_t)(1 * 16) * SEQ), o1);
        o2 = wmma_bf16(pa, *(const v16bf*)(vcol + (size_t)(2 * 16) * SEQ), o2);
        o3 = wmma_bf16(pa, *(const v16bf*)(vcol + (size_t)(3 * 16) * SEQ), o3);
    }

#pragma unroll
    for (int r = 0; r < 8; ++r) {
        float inv = 1.0f / l[r];
        int row = b * SEQ + qt * 16 + half * 8 + r;
        size_t base = (size_t)row * HID + h * HD + l16;
        O[base +  0] = (__bf16)(o0[r] * inv);
        O[base + 16] = (__bf16)(o1[r] * inv);
        O[base + 32] = (__bf16)(o2[r] * inv);
        O[base + 48] = (__bf16)(o3[r] * inv);
    }
}

// ---------------- host launch ----------------

extern "C" void kernel_launch(void* const* d_in, const int* in_sizes, int n_in,
                              void* d_out, int out_size, void* d_ws, size_t ws_size,
                              hipStream_t stream) {
    const float* x  = (const float*)d_in[0];
    const float* Wq = (const float*)d_in[1];
    const float* bq = (const float*)d_in[2];
    const float* Wk = (const float*)d_in[3];
    const float* bk = (const float*)d_in[4];
    const float* Wv = (const float*)d_in[5];
    const float* bv = (const float*)d_in[6];
    const float* Wo = (const float*)d_in[7];
    const float* bo = (const float*)d_in[8];
    float* out = (float*)d_out;

    char* ws = (char*)d_ws;
    __bf16* xBf    = (__bf16*)(ws + 0);         // 8 MiB
    __bf16* qBf    = (__bf16*)(ws + 8388608);   // 8 MiB
    __bf16* kBf    = (__bf16*)(ws + 16777216);  // 2 MiB
    __bf16* vBf    = (__bf16*)(ws + 18874368);  // 2 MiB
    __bf16* vtBf   = (__bf16*)(ws + 20971520);  // 2 MiB
    __bf16* attnBf = (__bf16*)(ws + 23068672);  // 8 MiB
    __bf16* wqT    = (__bf16*)(ws + 31457280);  // 2 MiB
    __bf16* wkT    = (__bf16*)(ws + 33554432);  // 512 KiB
    __bf16* wvT    = (__bf16*)(ws + 34078720);  // 512 KiB
    __bf16* woT    = (__bf16*)(ws + 34603008);  // 2 MiB

    cvt_f32_to_bf16<<<(MROWS * HID + 255) / 256, 256, 0, stream>>>(x, xBf, MROWS * HID);
    cvt_transpose_bf16<<<(HID * HID + 255) / 256, 256, 0, stream>>>(Wq, wqT, HID, HID);
    cvt_transpose_bf16<<<(HID * KVDIM + 255) / 256, 256, 0, stream>>>(Wk, wkT, HID, KVDIM);
    cvt_transpose_bf16<<<(HID * KVDIM + 255) / 256, 256, 0, stream>>>(Wv, wvT, HID, KVDIM);
    cvt_transpose_bf16<<<(HID * HID + 255) / 256, 256, 0, stream>>>(Wo, woT, HID, HID);

    // projections: 128x128 block tiles, TDM-fed
    gemm_bf16_tdm<<<(MROWS / 128) * (HID / 128), 256, 0, stream>>>(
        xBf, wqT, bq, qBf, nullptr, MROWS, HID, HID);
    gemm_bf16_tdm<<<(MROWS / 128) * (KVDIM / 128), 256, 0, stream>>>(
        xBf, wkT, bk, kBf, nullptr, MROWS, KVDIM, HID);
    gemm_bf16_tdm<<<(MROWS / 128) * (KVDIM / 128), 256, 0, stream>>>(
        xBf, wvT, bv, vBf, nullptr, MROWS, KVDIM, HID);

    transpose_v_kernel<<<(BATCH * NKV * HD * SEQ + 255) / 256, 256, 0, stream>>>(vBf, vtBf);

    gqa_attn_wmma<<<4096 / 8, 256, 0, stream>>>(qBf, kBf, vtBf, attnBf);

    gemm_bf16_tdm<<<(MROWS / 128) * (HID / 128), 256, 0, stream>>>(
        attnBf, woT, bo, nullptr, out, MROWS, HID, HID);
}
